// ViTSelfAttention3D_69621419868682
// MI455X (gfx1250) — matrix operand order repaired and checked by reading, not compiled, not run on hardware
//
#include <hip/hip_runtime.h>
#include <hip/hip_bf16.h>
#include <stdint.h>

// ---------------------------------------------------------------------------
// CDNA5 (gfx1250) ViT self-attention: QKV GEMM -> flash attention -> dense GEMM
// All matmuls on v_wmma_f32_16x16x32_bf16 (fp32 accumulate), intermediates bf16.
// K-tile staging uses global_load_async_to_lds_b32 (ASYNCcnt path).
// ---------------------------------------------------------------------------

typedef __attribute__((ext_vector_type(16))) __bf16 v16bf;
typedef __attribute__((ext_vector_type(8)))  float  v8f;

struct alignas(16) U4 { uint32_t x, y, z, w; };

union Frag {
  v16bf v;
  U4    q[2];
  uint32_t u[8];
};

__device__ __forceinline__ uint16_t f32_bf16(float f) {
  union { float f; uint32_t u; } c{f};
  uint32_t u = c.u;
  u += 0x7FFFu + ((u >> 16) & 1u);   // round-to-nearest-even
  return (uint16_t)(u >> 16);
}

__device__ __forceinline__ v8f wmma_bf16(const Frag& a, const Frag& b, v8f c) {
  // (neg_a, A, neg_b, B, c_mod, C, reuse_a, reuse_b)
  return __builtin_amdgcn_wmma_f32_16x16x32_bf16(false, a.v, false, b.v,
                                                 (short)0, c, false, false);
}

// async global->LDS copy of one DWORD per lane (gfx1250, ASYNCcnt-tracked)
__device__ __forceinline__ void async_ld_b32(const void* gaddr, void* lds_generic) {
  uint32_t lds = (uint32_t)(uintptr_t)lds_generic;  // flat->LDS offset: addr[31:0]
  asm volatile("global_load_async_to_lds_b32 %0, %1, off"
               :
               : "v"(lds), "v"((uint64_t)(uintptr_t)gaddr)
               : "memory");
}
__device__ __forceinline__ void wait_asynccnt0() {
  asm volatile("s_wait_asynccnt 0x0" ::: "memory");
}

// ---------------------------------------------------------------------------
// f32 -> bf16 pack (2 elements / thread, 4B stores)
// ---------------------------------------------------------------------------
__global__ void cvt_f32_bf16(const float* __restrict__ in,
                             uint16_t* __restrict__ out, long long n2) {
  long long i = (long long)blockIdx.x * blockDim.x + threadIdx.x;
  if (i < n2) {
    float a = in[2 * i], b = in[2 * i + 1];
    uint32_t pk = (uint32_t)f32_bf16(a) | ((uint32_t)f32_bf16(b) << 16);
    *(uint32_t*)(out + 2 * i) = pk;
  }
}

// ---------------------------------------------------------------------------
// Tiled bf16 WMMA GEMM: C[M,N] = A[M,K] * B[K,N] + bias[N]
// Block tile 128x128, K-step 32, 256 threads = 8 waves (wave32),
// wave grid 2(M) x 4(N); each wave: 4x2 accumulators of 16x16.
// ---------------------------------------------------------------------------
template <bool OUT_F32>
__global__ __launch_bounds__(256)
void gemm_bf16(const uint16_t* __restrict__ A, const uint16_t* __restrict__ B,
               const float* __restrict__ bias, void* __restrict__ Out,
               int M, int N, int K) {
  __shared__ __align__(16) uint16_t As[128 * 40];  // pad: stride 80B (5*16)
  __shared__ __align__(16) uint16_t Bs[128 * 40];

  const int tid  = threadIdx.x;
  const int w    = tid >> 5, lane = tid & 31;
  const int half = lane >> 4, l15 = lane & 15;
  const int wm   = w & 1, wn = w >> 1;
  const int m0   = blockIdx.y * 128, n0 = blockIdx.x * 128;

  v8f acc[4][2];
#pragma unroll
  for (int i = 0; i < 4; ++i)
#pragma unroll
    for (int j = 0; j < 2; ++j) acc[i][j] = {};

  for (int kk = 0; kk < K; kk += 32) {
    // stage A tile 128x32 via async global->LDS (row-major, no transpose needed)
#pragma unroll
    for (int it = 0; it < 8; ++it) {
      int c = tid + it * 256;
      int row = c >> 4, kp = (c & 15) << 1;
      async_ld_b32(A + (size_t)(m0 + row) * K + kk + kp, As + row * 40 + kp);
    }
    // stage B tile 32x128 transposed into Bs[n][k], coalesced along N
#pragma unroll
    for (int it = 0; it < 8; ++it) {
      int c = tid + it * 256;
      int kr = c >> 6, np = (c & 63) << 1;
      uint32_t v = *(const uint32_t*)(B + (size_t)(kk + kr) * N + n0 + np);
      Bs[np * 40 + kr]       = (uint16_t)(v & 0xFFFFu);
      Bs[(np + 1) * 40 + kr] = (uint16_t)(v >> 16);
    }
    if (kk + 32 < K) {  // global_prefetch_b8 of next tiles
      __builtin_prefetch(A + (size_t)(m0 + (tid >> 1)) * K + kk + 32, 0, 0);
      __builtin_prefetch(B + (size_t)(kk + 32 + (tid >> 3)) * N + n0 + ((tid & 7) << 4), 0, 0);
    }
    wait_asynccnt0();
    __syncthreads();

    // A fragments: 16-bit A layout (element i -> K = (i&7) + 8*half + 16*(i>>3))
    Frag a[4];
#pragma unroll
    for (int i = 0; i < 4; ++i) {
      const uint16_t* p = As + (wm * 64 + i * 16 + l15) * 40;
      a[i].q[0] = *(const U4*)(p + 8 * half);
      a[i].q[1] = *(const U4*)(p + 16 + 8 * half);
    }
    // B fragments: same K order as A (element i -> K = (i&7) + 8*half + 16*(i>>3))
    Frag bfr[2];
#pragma unroll
    for (int j = 0; j < 2; ++j) {
      const uint16_t* p = Bs + (wn * 32 + j * 16 + l15) * 40;
      bfr[j].q[0] = *(const U4*)(p + 8 * half);
      bfr[j].q[1] = *(const U4*)(p + 16 + 8 * half);
    }
#pragma unroll
    for (int i = 0; i < 4; ++i)
#pragma unroll
      for (int j = 0; j < 2; ++j)
        acc[i][j] = wmma_bf16(a[i], bfr[j], acc[i][j]);
    __syncthreads();
  }

  // epilogue: C layout lane -> N=l15, VGPR r -> M = r + 8*half
#pragma unroll
  for (int j = 0; j < 2; ++j) {
    int col = n0 + wn * 32 + j * 16 + l15;
    float bv = bias[col];
#pragma unroll
    for (int i = 0; i < 4; ++i) {
#pragma unroll
      for (int r = 0; r < 8; ++r) {
        int row = m0 + wm * 64 + i * 16 + r + 8 * half;
        float v = acc[i][j][r] + bv;
        if (OUT_F32)
          ((float*)Out)[(size_t)row * N + col] = v;
        else
          ((uint16_t*)Out)[(size_t)row * N + col] = f32_bf16(v);
      }
    }
  }
}

// ---------------------------------------------------------------------------
// Flash attention over fused QKV (bf16, layout [token][head*192 + {q|k|v}*64]).
// Grid: (B*HEADS, S/128). Block 256 threads = 8 waves; wave owns 16 q rows.
// ---------------------------------------------------------------------------
__global__ __launch_bounds__(256)
void flash_attn(const uint16_t* __restrict__ QKV, uint16_t* __restrict__ CTX) {
  __shared__ __align__(16) uint16_t Kl[64 * 72];       // K tile [kv][d], pad 72
  __shared__ __align__(16) uint16_t Vl[64 * 72];       // V tile transposed [d][kv]
  __shared__ __align__(16) uint16_t Pl[8 * 16 * 72];   // per-wave P staging

  const int tid  = threadIdx.x;
  const int w    = tid >> 5, lane = tid & 31;
  const int half = lane >> 4, l15 = lane & 15;
  const int b    = blockIdx.x >> 4, h = blockIdx.x & 15;
  const int q0   = blockIdx.y * 128 + w * 16;

  // Q fragments (A layout) straight from global: rows q0+l15, d 0..63.
  // Pre-scale by 1/sqrt(64) = 0.125 (exact in bf16: exponent shift).
  Frag qf[2];
  {
    const uint16_t* qp = QKV + ((size_t)(b * 2048 + q0 + l15)) * 3072 + h * 192;
#pragma unroll
    for (int c = 0; c < 2; ++c) {
      qf[c].q[0] = *(const U4*)(qp + c * 32 + 8 * half);
      qf[c].q[1] = *(const U4*)(qp + c * 32 + 16 + 8 * half);
#pragma unroll
      for (int e = 0; e < 8; ++e) {
        uint32_t pair = qf[c].u[e];
        float lo = __uint_as_float((pair & 0x0000FFFFu) << 16) * 0.125f;
        float hi = __uint_as_float(pair & 0xFFFF0000u) * 0.125f;
        qf[c].u[e] = (uint32_t)f32_bf16(lo) | ((uint32_t)f32_bf16(hi) << 16);
      }
    }
  }

  v8f ctx[4];
#pragma unroll
  for (int t = 0; t < 4; ++t) ctx[t] = {};
  float run_m[8], run_l[8];
#pragma unroll
  for (int r = 0; r < 8; ++r) { run_m[r] = -3.402823466e38f; run_l[r] = 0.f; }

  uint16_t* pw = Pl + w * 16 * 72;

  for (int kv0 = 0; kv0 < 2048; kv0 += 64) {
    // K tile: async global->LDS (no VGPR round-trip, ASYNCcnt-tracked).
    // V tile: load + transpose into Vl[d][kv] (d16_hi stores).
#pragma unroll
    for (int it = 0; it < 8; ++it) {
      int c = tid + it * 256;
      int row = c >> 5, dp = (c & 31) << 1;
      const uint16_t* src = QKV + ((size_t)(b * 2048 + kv0 + row)) * 3072 + h * 192;
      async_ld_b32(src + 64 + dp, Kl + row * 72 + dp);
      uint32_t vvv = *(const uint32_t*)(src + 128 + dp);
      Vl[dp * 72 + row]       = (uint16_t)(vvv & 0xFFFFu);
      Vl[(dp + 1) * 72 + row] = (uint16_t)(vvv >> 16);
    }
    wait_asynccnt0();
    __syncthreads();

    // S = (Q/8) @ K^T : 4 column subtiles of 16 kv, K-dim 64 = 2 WMMAs each
    v8f s[4];
#pragma unroll
    for (int j = 0; j < 4; ++j) {
      v8f accs = {};
#pragma unroll
      for (int c = 0; c < 2; ++c) {
        Frag kb;
        const uint16_t* p = Kl + (j * 16 + l15) * 72 + c * 32;
        kb.q[0] = *(const U4*)(p + 8 * half);
        kb.q[1] = *(const U4*)(p + 16 + 8 * half);
        accs = wmma_bf16(qf[c], kb, accs);
      }
      s[j] = accs;
    }

    // online softmax: row m = r + 8*half lives across the lane's 16-lane half
#pragma unroll
    for (int r = 0; r < 8; ++r) {
      float mx = fmaxf(fmaxf(s[0][r], s[1][r]), fmaxf(s[2][r], s[3][r]));
      mx = fmaxf(mx, __shfl_xor(mx, 1, 32));
      mx = fmaxf(mx, __shfl_xor(mx, 2, 32));
      mx = fmaxf(mx, __shfl_xor(mx, 4, 32));
      mx = fmaxf(mx, __shfl_xor(mx, 8, 32));
      float m_new = fmaxf(run_m[r], mx);
      float alpha = __expf(run_m[r] - m_new);
      float rs = 0.f;
#pragma unroll
      for (int j = 0; j < 4; ++j) {
        float p = __expf(s[j][r] - m_new);
        s[j][r] = p;
        rs += p;
      }
      rs += __shfl_xor(rs, 1, 32);
      rs += __shfl_xor(rs, 2, 32);
      rs += __shfl_xor(rs, 4, 32);
      rs += __shfl_xor(rs, 8, 32);
      run_l[r] = run_l[r] * alpha + rs;
      run_m[r] = m_new;
#pragma unroll
      for (int t = 0; t < 4; ++t) ctx[t][r] *= alpha;
    }

    // C-layout P -> LDS as truncated bf16 (backend emits ds_store_b16_d16_hi,
    // zero VALU; truncation of softmax probabilities is accuracy-neutral here)
#pragma unroll
    for (int j = 0; j < 4; ++j)
#pragma unroll
      for (int r = 0; r < 8; ++r)
        pw[(r + 8 * half) * 72 + j * 16 + l15] =
            (uint16_t)(__float_as_uint(s[j][r]) >> 16);

    Frag pf[2];
#pragma unroll
    for (int c = 0; c < 2; ++c) {
      const uint16_t* p = pw + l15 * 72 + c * 32;
      pf[c].q[0] = *(const U4*)(p + 8 * half);
      pf[c].q[1] = *(const U4*)(p + 16 + 8 * half);
    }

    // ctx += P @ V  (V as B fragments from transposed tile)
#pragma unroll
    for (int t = 0; t < 4; ++t) {
#pragma unroll
      for (int c = 0; c < 2; ++c) {
        Frag vb;
        const uint16_t* p = Vl + (t * 16 + l15) * 72 + c * 32;
        vb.q[0] = *(const U4*)(p + 8 * half);
        vb.q[1] = *(const U4*)(p + 16 + 8 * half);
        ctx[t] = wmma_bf16(pf[c], vb, ctx[t]);
      }
    }
    __syncthreads();
  }

  // normalize and store ctx as [token][h*64 + d] (bf16)
#pragma unroll
  for (int t = 0; t < 4; ++t) {
#pragma unroll
    for (int r = 0; r < 8; ++r) {
      int tok = b * 2048 + q0 + r + 8 * half;
      int col = h * 64 + t * 16 + l15;
      CTX[(size_t)tok * 1024 + col] = f32_bf16(ctx[t][r] * (1.f / run_l[r]));
    }
  }
}

// ---------------------------------------------------------------------------
extern "C" void kernel_launch(void* const* d_in, const int* in_sizes, int n_in,
                              void* d_out, int out_size, void* d_ws, size_t ws_size,
                              hipStream_t stream) {
  const float* X  = (const float*)d_in[0];   // [2,2048,1024]
  const float* Wq = (const float*)d_in[1];   // [1024,3072]
  const float* bq = (const float*)d_in[2];   // [3072]
  const float* Wd = (const float*)d_in[3];   // [1024,1024]
  const float* bd = (const float*)d_in[4];   // [1024]

  const long long nX  = 4096ll * 1024;
  const long long nWq = 1024ll * 3072;
  const long long nWd = 1024ll * 1024;
  const long long nQKV = 4096ll * 3072;
  const long long nCTX = 4096ll * 1024;

  uint8_t* ws = (uint8_t*)d_ws;
  uint16_t* Xb   = (uint16_t*)ws; ws += (size_t)nX * 2;
  uint16_t* Wqb  = (uint16_t*)ws; ws += (size_t)nWq * 2;
  uint16_t* Wdb  = (uint16_t*)ws; ws += (size_t)nWd * 2;
  uint16_t* QKVb = (uint16_t*)ws; ws += (size_t)nQKV * 2;
  uint16_t* CTXb = (uint16_t*)ws; ws += (size_t)nCTX * 2;

  auto cvt_blocks = [](long long n) { return (unsigned)((n / 2 + 255) / 256); };
  cvt_f32_bf16<<<cvt_blocks(nX),  256, 0, stream>>>(X,  Xb,  nX / 2);
  cvt_f32_bf16<<<cvt_blocks(nWq), 256, 0, stream>>>(Wq, Wqb, nWq / 2);
  cvt_f32_bf16<<<cvt_blocks(nWd), 256, 0, stream>>>(Wd, Wdb, nWd / 2);

  // QKV projection: [4096,1024] @ [1024,3072] -> bf16 QKV
  gemm_bf16<false><<<dim3(3072 / 128, 4096 / 128), 256, 0, stream>>>(
      Xb, Wqb, bq, QKVb, 4096, 3072, 1024);

  // flash attention over all (b, head, q-tile)
  flash_attn<<<dim3(32, 16), 256, 0, stream>>>(QKVb, CTXb);

  // dense projection: [4096,1024] @ [1024,1024] -> f32 out
  gemm_bf16<true><<<dim3(1024 / 128, 4096 / 128), 256, 0, stream>>>(
      CTXb, Wdb, bd, d_out, 4096, 1024, 1024);
}
